// Model_82051055223385
// MI455X (gfx1250) — matrix operand order repaired and checked by reading, compiled, not run on hardware
//
#include <hip/hip_runtime.h>
#include <math.h>

// ---------------------------------------------------------------------------
// CDNA5 / gfx1250, wave32. Heavy matmuls via v_wmma_f32_16x16x32_bf16,
// enc-panel staging via TDM (tensor_load_to_lds + s_wait_tensorcnt).
// ---------------------------------------------------------------------------

typedef __bf16 bf16_t;
typedef __attribute__((ext_vector_type(16))) __bf16 v16bf;
typedef __attribute__((ext_vector_type(8)))  __bf16 v8bf;
typedef __attribute__((ext_vector_type(8)))  float  v8f;
typedef __attribute__((ext_vector_type(4)))  unsigned int u32x4;
typedef __attribute__((ext_vector_type(8)))  int i32x8;
typedef __attribute__((ext_vector_type(4)))  int i32x4;

constexpr int kN   = 8192;   // cells
constexpr int kG   = 512;    // genes
constexpr int kDD  = 512;    // denoise hidden
constexpr int kDE1 = 256;    // embed hidden
constexpr int kDE  = 32;     // embed dim (== one WMMA K step)
constexpr int kDQ  = 64;     // quality hidden
constexpr int kH   = 4;      // heads

#if __has_builtin(__builtin_amdgcn_tensor_load_to_lds)
#define HAVE_TDM 1
#else
#define HAVE_TDM 0
#endif

// --- WMMA fragment helpers (ISA 7.12.2 layouts, wave32) --------------------
// A frag: lane holds row (lane%16); K = {8*half .. +7} and {16+8*half .. +7}
__device__ inline v16bf frag_a(const bf16_t* row, int hf) {
  v8bf lo = *(const v8bf*)(row + 8 * hf);
  v8bf hi = *(const v8bf*)(row + 16 + 8 * hf);
  v16bf f;
#pragma unroll
  for (int i = 0; i < 8; ++i) { f[i] = lo[i]; f[i + 8] = hi[i]; }
  return f;
}
// B frag from B^T (row = output column): K = {8*half .. +7} and {16+8*half .. +7}
__device__ inline v16bf frag_b(const bf16_t* row, int hf) {
  v8bf lo = *(const v8bf*)(row + 8 * hf);
  v8bf hi = *(const v8bf*)(row + 16 + 8 * hf);
  v16bf f;
#pragma unroll
  for (int i = 0; i < 8; ++i) { f[i] = lo[i]; f[i + 8] = hi[i]; }
  return f;
}
__device__ inline v8f wmma_bf16(v16bf a, v16bf b, v8f c) {
  return __builtin_amdgcn_wmma_f32_16x16x32_bf16(false, a, false, b,
                                                 (short)0, c, false, false);
}
__device__ inline float red_max16(float x) {
#pragma unroll
  for (int m = 1; m < 16; m <<= 1) x = fmaxf(x, __shfl_xor(x, m, 32));
  return x;
}
__device__ inline float red_sum16(float x) {
#pragma unroll
  for (int m = 1; m < 16; m <<= 1) x += __shfl_xor(x, m, 32);
  return x;
}

// --- TDM staging: 2KB contiguous (32 enc rows x 32 bf16) global -> LDS -----
// D# per CDNA5 ISA: group0 = {count, lds_addr, global_addr,
// type=2}; group1 = {data_size=3 (8B units), tensor_dim0=256, tile_dim0=256,
// tile_dim1=1, stride0=256}; groups 2/3 zero (<=2D tensor).
// This toolchain exposes the 6-arg builtin (extra int32x8 group before cpol).
__device__ inline void stage_enc(const bf16_t* src, bf16_t* dstLds, int lane) {
#if HAVE_TDM
  unsigned long long ga = (unsigned long long)(uintptr_t)src;
  u32x4 g0;
  g0[0] = 1u;                                            // count=1, user mode
  g0[1] = (unsigned)(uintptr_t)dstLds;                   // LDS byte address
  g0[2] = (unsigned)(ga & 0xffffffffu);                  // global addr lo
  g0[3] = (unsigned)((ga >> 32) & 0x01ffffffu) | (2u << 30);  // hi | type=2
  i32x8 g1;
  g1[0] = 0x00030000;   // data_size=3 (8B), no multicast mask
  g1[1] = 0x01000000;   // tensor_dim0 = 256 (bits [63:48])
  g1[2] = 0x00010000;   // tensor_dim1 = 1   (bits [95:80])
  g1[3] = 0x01000000;   // tile_dim0  = 256  (bits [127:112])
  g1[4] = 0x00000001;   // tile_dim1  = 1
  g1[5] = 256;          // tensor_dim0_stride = 256 units
  g1[6] = 0; g1[7] = 0;
  i32x4 z4 = {0, 0, 0, 0};
  i32x8 z8;
#pragma unroll
  for (int i = 0; i < 8; ++i) z8[i] = 0;
  __builtin_amdgcn_tensor_load_to_lds(g0, g1, z4, z4, z8, 0);
  (void)lane;
#else
  const v8bf* s = (const v8bf*)src + lane * 4;           // 64B per lane
  v8bf* d = (v8bf*)dstLds + lane * 4;
  d[0] = s[0]; d[1] = s[1]; d[2] = s[2]; d[3] = s[3];
#endif
}
__device__ inline void stage_wait() {
#if HAVE_TDM
#if __has_builtin(__builtin_amdgcn_s_wait_tensorcnt)
  __builtin_amdgcn_s_wait_tensorcnt(0);
#else
  asm volatile("s_wait_tensorcnt 0x0" ::: "memory");
#endif
#endif
}

// --- small utility kernels -------------------------------------------------
__global__ void k_f32_to_bf16(bf16_t* dst, const float* src, int n) {
  int i = blockIdx.x * blockDim.x + threadIdx.x;
  if (i < n) dst[i] = (bf16_t)src[i];
}
__global__ void k_transpose_bf16(bf16_t* dst, const float* src, int R, int C) {
  int i = blockIdx.x * blockDim.x + threadIdx.x;
  if (i >= R * C) return;
  int r = i / C, c = i % C;
  dst[(size_t)c * R + r] = (bf16_t)src[i];
}
__global__ void k_quality_sq(const bf16_t* Hq, const float* qW2, const float* qb2,
                             const float* encf, float* qual, float* sq) {
  int n = blockIdx.x * blockDim.x + threadIdx.x;
  if (n >= kN) return;
  float q = qb2[0];
#pragma unroll 8
  for (int k = 0; k < kDQ; ++k) q += (float)Hq[(size_t)n * kDQ + k] * qW2[k];
  qual[n] = q;
  float s = 0.f;
#pragma unroll
  for (int d = 0; d < kDE; ++d) { float e = encf[(size_t)n * kDE + d]; s += e * e; }
  sq[n] = s;
}
__global__ void k_final_add(float* out, const float* sm, const float* inter, int n) {
  int i = blockIdx.x * blockDim.x + threadIdx.x;
  if (i < n) out[i] = sm[i] + inter[i];
}

// --- generic tiled WMMA GEMM: C = act(A[MxK] * Bt[NxK]^T + bias) (+opts) ---
template <int ACT_TANH, int RES, int ACC, int OF32, int OBF, int OBT>
__global__ __launch_bounds__(256)
void k_gemm(const bf16_t* __restrict__ A, const bf16_t* __restrict__ Bt,
            const float* __restrict__ bias, const float* __restrict__ resid,
            float* __restrict__ accum, float* __restrict__ outF,
            bf16_t* __restrict__ outB, bf16_t* __restrict__ outBT,
            int M, int N, int K) {
  const int tid = threadIdx.x, wave = tid >> 5, lane = tid & 31;
  const int hf = lane >> 4, nn = lane & 15;
  const int wm = wave & 3, wn = wave >> 2;
  const int row0 = blockIdx.y * 64 + wm * 16;
  const int col0 = blockIdx.x * 32 + wn * 16;
  const bf16_t* Ar = A + (size_t)(row0 + nn) * K;
  const bf16_t* Br = Bt + (size_t)(col0 + nn) * K;
  v8f acc = {};
  for (int kb = 0; kb < K; kb += 32) {
    if (kb + 64 <= K) {  // global_prefetch_b8 warmup
      __builtin_prefetch(Ar + kb + 32, 0, 0);
      __builtin_prefetch(Br + kb + 32, 0, 0);
    }
    v16bf a = frag_a(Ar + kb, hf);
    v16bf b = frag_b(Br + kb, hf);
    acc = wmma_bf16(a, b, acc);
  }
  const int col = col0 + nn;
  const float bv = bias ? bias[col] : 0.f;
#pragma unroll
  for (int r = 0; r < 8; ++r) {
    const int row = row0 + 8 * hf + r;
    const size_t idx = (size_t)row * N + col;
    float v = acc[r] + bv;
    if constexpr (ACT_TANH) v = tanhf(v);
    if constexpr (RES) v += resid[idx];
    if constexpr (ACC) v += accum[idx];
    if constexpr (OF32) outF[idx] = v;
    if constexpr (OBF) outB[idx] = (bf16_t)v;
    if constexpr (OBT) outBT[(size_t)col * M + row] = (bf16_t)v;
  }
}

// --- fused smooth: softmax(-cdist(enc)+qual^T) @ denoised ------------------
// 8 waves: 2 rowgroups x 16 rows; in each rowgroup wave gs==0 computes the
// P tile ONCE (shared via LDS), all 4 waves run P@V on 128-gene slices.
// enc j-panels double-buffered in LDS via TDM.
__global__ __launch_bounds__(256)
void k_flash_smooth(const bf16_t* __restrict__ enc, const bf16_t* __restrict__ denT,
                    const float* __restrict__ sq, const float* __restrict__ qual,
                    float* __restrict__ outS, bf16_t* __restrict__ smT) {
  __shared__ bf16_t ldsE[2][32 * kDE];   // staged enc chunk (double buffer)
  __shared__ bf16_t ldsP[2][16 * 32];    // probability tile per rowgroup
  __shared__ float  ldsF[2][16];         // per-row rescale factors
  __shared__ float  ldsS[2][16];         // final row sums
  const int tid = threadIdx.x, wave = tid >> 5, lane = tid & 31;
  const int hf = lane >> 4, nn = lane & 15;
  const int rg = wave >> 2, gs = wave & 3;
  const int row0 = blockIdx.x * 32 + rg * 16;
  const int g0 = gs * 128;

  const v16bf aq = frag_a(enc + (size_t)(row0 + nn) * kDE, hf);
  float sqrow[8], rowm[8], rsum[8];
#pragma unroll
  for (int r = 0; r < 8; ++r) {
    sqrow[r] = sq[row0 + 8 * hf + r];
    rowm[r] = -INFINITY; rsum[r] = 0.f;
  }
  v8f acc[8];
#pragma unroll
  for (int t = 0; t < 8; ++t)
#pragma unroll
    for (int r = 0; r < 8; ++r) acc[t][r] = 0.f;

  if (wave == 0) { stage_enc(enc, ldsE[0], lane); stage_wait(); }
  __syncthreads();

  for (int j0 = 0; j0 < kN; j0 += 32) {
    const int cur = (j0 >> 5) & 1;
    const bool more = (j0 + 32) < kN;
    if (wave == 0 && more)
      stage_enc(enc + (size_t)(j0 + 32) * kDE, ldsE[cur ^ 1], lane);

    if (gs == 0) {                         // one wave builds P for the rowgroup
      v8f z = {};
      v16bf b0 = frag_b(ldsE[cur] + nn * kDE, hf);
      v16bf b1 = frag_b(ldsE[cur] + (16 + nn) * kDE, hf);
      v8f d0 = wmma_bf16(aq, b0, z);
      v8f d1 = wmma_bf16(aq, b1, z);
      const float sqc0 = sq[j0 + nn], qc0 = qual[j0 + nn];
      const float sqc1 = sq[j0 + 16 + nn], qc1 = qual[j0 + 16 + nn];
#pragma unroll
      for (int r = 0; r < 8; ++r) {
        float t0 = sqrow[r] + sqc0 - 2.f * d0[r];
        float t1 = sqrow[r] + sqc1 - 2.f * d1[r];
        float s0 = qc0 - __builtin_amdgcn_sqrtf(fmaxf(t0, 0.f));  // raw v_sqrt
        float s1 = qc1 - __builtin_amdgcn_sqrtf(fmaxf(t1, 0.f));
        float cm = red_max16(fmaxf(s0, s1));
        float nm = fmaxf(rowm[r], cm);
        float fac = __expf(rowm[r] - nm);
        float p0 = __expf(s0 - nm), p1 = __expf(s1 - nm);
        rsum[r] = rsum[r] * fac + red_sum16(p0 + p1);
        rowm[r] = nm;
        if (nn == 0) ldsF[rg][8 * hf + r] = fac;
        ldsP[rg][(8 * hf + r) * 32 + nn] = (bf16_t)p0;
        ldsP[rg][(8 * hf + r) * 32 + 16 + nn] = (bf16_t)p1;
      }
    }
    __syncthreads();

    v16bf ap = frag_a(&ldsP[rg][nn * 32], hf);     // shared P tile -> A frag
#pragma unroll
    for (int r = 0; r < 8; ++r) {
      const float fac = ldsF[rg][8 * hf + r];
#pragma unroll
      for (int t = 0; t < 8; ++t) acc[t][r] *= fac;
    }
#pragma unroll
    for (int t = 0; t < 8; ++t) {                  // P @ denoised
      v16bf bv = frag_b(denT + (size_t)(g0 + t * 16 + nn) * kN + j0, hf);
      acc[t] = wmma_bf16(ap, bv, acc[t]);
    }
    if (wave == 0 && more) stage_wait();           // next enc panel landed
    __syncthreads();
  }

  if (gs == 0 && nn == 0) {
#pragma unroll
    for (int r = 0; r < 8; ++r) ldsS[rg][8 * hf + r] = rsum[r];
  }
  __syncthreads();
#pragma unroll
  for (int r = 0; r < 8; ++r) {
    const int row = row0 + 8 * hf + r;
    const float inv = __builtin_amdgcn_rcpf(ldsS[rg][8 * hf + r]);
#pragma unroll
    for (int t = 0; t < 8; ++t) {
      const int g = g0 + t * 16 + nn;
      float v = acc[t][r] * inv;
      outS[(size_t)row * kG + g] = v;
      smT[(size_t)g * kN + row] = (bf16_t)v;
    }
  }
}

// --- fused interact head: t_h = sigmoid(tenc @ enc^T) @ smoothed -----------
__global__ __launch_bounds__(256)
void k_flash_interact(const bf16_t* __restrict__ tenc, const bf16_t* __restrict__ enc,
                      const bf16_t* __restrict__ smT, bf16_t* __restrict__ tout,
                      float scale) {
  __shared__ bf16_t ldsE[2][32 * kDE];
  __shared__ bf16_t ldsP[2][16 * 32];
  const int tid = threadIdx.x, wave = tid >> 5, lane = tid & 31;
  const int hf = lane >> 4, nn = lane & 15;
  const int rg = wave >> 2, gs = wave & 3;
  const int row0 = blockIdx.x * 32 + rg * 16;
  const int g0 = gs * 128;

  const v16bf aq = frag_a(tenc + (size_t)(row0 + nn) * kDE, hf);
  v8f acc[8];
#pragma unroll
  for (int t = 0; t < 8; ++t)
#pragma unroll
    for (int r = 0; r < 8; ++r) acc[t][r] = 0.f;

  if (wave == 0) { stage_enc(enc, ldsE[0], lane); stage_wait(); }
  __syncthreads();

  for (int j0 = 0; j0 < kN; j0 += 32) {
    const int cur = (j0 >> 5) & 1;
    const bool more = (j0 + 32) < kN;
    if (wave == 0 && more)
      stage_enc(enc + (size_t)(j0 + 32) * kDE, ldsE[cur ^ 1], lane);

    if (gs == 0) {
      v8f z = {};
      v16bf b0 = frag_b(ldsE[cur] + nn * kDE, hf);
      v16bf b1 = frag_b(ldsE[cur] + (16 + nn) * kDE, hf);
      v8f d0 = wmma_bf16(aq, b0, z);
      v8f d1 = wmma_bf16(aq, b1, z);
#pragma unroll
      for (int r = 0; r < 8; ++r) {
        float p0 = __builtin_amdgcn_rcpf(1.f + __expf(-d0[r]));  // sigmoid
        float p1 = __builtin_amdgcn_rcpf(1.f + __expf(-d1[r]));
        ldsP[rg][(8 * hf + r) * 32 + nn] = (bf16_t)p0;
        ldsP[rg][(8 * hf + r) * 32 + 16 + nn] = (bf16_t)p1;
      }
    }
    __syncthreads();

    v16bf ap = frag_a(&ldsP[rg][nn * 32], hf);
#pragma unroll
    for (int t = 0; t < 8; ++t) {
      v16bf bv = frag_b(smT + (size_t)(g0 + t * 16 + nn) * kN + j0, hf);
      acc[t] = wmma_bf16(ap, bv, acc[t]);
    }
    if (wave == 0 && more) stage_wait();
    __syncthreads();
  }
#pragma unroll
  for (int r = 0; r < 8; ++r) {
    const int row = row0 + 8 * hf + r;
#pragma unroll
    for (int t = 0; t < 8; ++t) {
      const int g = g0 + t * 16 + nn;
      tout[(size_t)row * kG + g] = (bf16_t)(acc[t][r] * scale);  // fold 1/N
    }
  }
}

// ---------------------------------------------------------------------------
extern "C" void kernel_launch(void* const* d_in, const int* in_sizes, int n_in,
                              void* d_out, int out_size, void* d_ws, size_t ws_size,
                              hipStream_t stream) {
  (void)in_sizes; (void)n_in; (void)out_size; (void)ws_size;
  const float* raw = (const float*)d_in[0];
  const float* dW1 = (const float*)d_in[1];  const float* db1 = (const float*)d_in[2];
  const float* dW2 = (const float*)d_in[3];  const float* db2 = (const float*)d_in[4];
  const float* eW1 = (const float*)d_in[5];  const float* eb1 = (const float*)d_in[6];
  const float* eW2 = (const float*)d_in[7];  const float* eb2 = (const float*)d_in[8];
  const float* qW1 = (const float*)d_in[9];  const float* qb1 = (const float*)d_in[10];
  const float* qW2 = (const float*)d_in[11]; const float* qb2 = (const float*)d_in[12];
  const float* trf = (const float*)d_in[13]; const float* grs = (const float*)d_in[14];

  const size_t NG = (size_t)kN * kG;
  float* out_den = (float*)d_out;
  float* out_sm  = out_den + NG;
  float* out_fin = out_sm + NG;

  char* p = (char*)d_ws;
  auto take = [&](size_t bytes) -> void* {
    void* r = (void*)p; p += (bytes + 255) & ~(size_t)255; return r;
  };
  bf16_t* raw_b = (bf16_t*)take(NG * 2);
  bf16_t* H1    = (bf16_t*)take(NG * 2);
  bf16_t* denT  = (bf16_t*)take(NG * 2);
  bf16_t* H2    = (bf16_t*)take((size_t)kN * kDE1 * 2);
  float*  encf  = (float*) take((size_t)kN * kDE * 4);
  bf16_t* encb  = (bf16_t*)take((size_t)kN * kDE * 2);
  bf16_t* Hq    = (bf16_t*)take((size_t)kN * kDQ * 2);
  float*  qual  = (float*) take((size_t)kN * 4);
  float*  sq    = (float*) take((size_t)kN * 4);
  bf16_t* smT   = (bf16_t*)take(NG * 2);
  bf16_t* tenc  = (bf16_t*)take((size_t)kN * kDE * 2);
  bf16_t* th    = (bf16_t*)take(NG * 2);
  float*  inter = (float*) take(NG * 4);
  bf16_t* dW1t  = (bf16_t*)take((size_t)kG * kDD * 2);
  bf16_t* dW2t  = (bf16_t*)take((size_t)kDD * kG * 2);
  bf16_t* eW1t  = (bf16_t*)take((size_t)kG * kDE1 * 2);
  bf16_t* eW2t  = (bf16_t*)take((size_t)kDE1 * kDE * 2);
  bf16_t* qW1t  = (bf16_t*)take((size_t)kG * kDQ * 2);
  bf16_t* Tt    = (bf16_t*)take((size_t)kH * kDE * kDE * 2);
  bf16_t* grT   = (bf16_t*)take((size_t)kH * kG * kG * 2);

  auto cdiv = [](long a, long b) { return (unsigned)((a + b - 1) / b); };
  const dim3 blk(256);

  k_f32_to_bf16<<<cdiv(NG, 256), blk, 0, stream>>>(raw_b, raw, (int)NG);
  k_transpose_bf16<<<cdiv((long)kG * kDD, 256), blk, 0, stream>>>(dW1t, dW1, kG, kDD);
  k_transpose_bf16<<<cdiv((long)kDD * kG, 256), blk, 0, stream>>>(dW2t, dW2, kDD, kG);
  k_transpose_bf16<<<cdiv((long)kG * kDE1, 256), blk, 0, stream>>>(eW1t, eW1, kG, kDE1);
  k_transpose_bf16<<<cdiv((long)kDE1 * kDE, 256), blk, 0, stream>>>(eW2t, eW2, kDE1, kDE);
  k_transpose_bf16<<<cdiv((long)kG * kDQ, 256), blk, 0, stream>>>(qW1t, qW1, kG, kDQ);
  for (int h = 0; h < kH; ++h) {
    k_transpose_bf16<<<cdiv((long)kDE * kDE, 256), blk, 0, stream>>>(
        Tt + (size_t)h * kDE * kDE, trf + (size_t)h * kDE * kDE, kDE, kDE);
    k_transpose_bf16<<<cdiv((long)kG * kG, 256), blk, 0, stream>>>(
        grT + (size_t)h * kG * kG, grs + (size_t)h * kG * kG, kG, kG);
  }

  // CellDenoise
  k_gemm<1, 0, 0, 0, 1, 0><<<dim3(kDD / 32, kN / 64), blk, 0, stream>>>(
      raw_b, dW1t, db1, nullptr, nullptr, nullptr, H1, nullptr, kN, kDD, kG);
  k_gemm<0, 1, 0, 1, 0, 1><<<dim3(kG / 32, kN / 64), blk, 0, stream>>>(
      H1, dW2t, db2, raw, nullptr, out_den, nullptr, denT, kN, kG, kDD);
  // CellEmbed
  k_gemm<1, 0, 0, 0, 1, 0><<<dim3(kDE1 / 32, kN / 64), blk, 0, stream>>>(
      raw_b, eW1t, eb1, nullptr, nullptr, nullptr, H2, nullptr, kN, kDE1, kG);
  k_gemm<0, 0, 0, 1, 1, 0><<<dim3(kDE / 32, kN / 64), blk, 0, stream>>>(
      H2, eW2t, eb2, nullptr, nullptr, encf, encb, nullptr, kN, kDE, kDE1);
  // CellQualify + |enc|^2
  k_gemm<1, 0, 0, 0, 1, 0><<<dim3(kDQ / 32, kN / 64), blk, 0, stream>>>(
      raw_b, qW1t, qb1, nullptr, nullptr, nullptr, Hq, nullptr, kN, kDQ, kG);
  k_quality_sq<<<cdiv(kN, 256), blk, 0, stream>>>(Hq, qW2, qb2, encf, qual, sq);

  // CellSmooth (flash softmax, TDM-staged enc panels)
  k_flash_smooth<<<kN / 32, blk, 0, stream>>>(encb, denT, sq, qual, out_sm, smT);

  // CellInteract heads
  for (int h = 0; h < kH; ++h) {
    k_gemm<0, 0, 0, 0, 1, 0><<<dim3(kDE / 32, kN / 64), blk, 0, stream>>>(
        encb, Tt + (size_t)h * kDE * kDE, nullptr, nullptr, nullptr, nullptr,
        tenc, nullptr, kN, kDE, kDE);
    k_flash_interact<<<kN / 32, blk, 0, stream>>>(tenc, encb, smT, th, 1.f / kN);
    if (h == 0)
      k_gemm<0, 0, 0, 1, 0, 0><<<dim3(kG / 32, kN / 64), blk, 0, stream>>>(
          th, grT, nullptr, nullptr, nullptr, inter, nullptr, nullptr, kN, kG, kG);
    else
      k_gemm<0, 0, 1, 1, 0, 0><<<dim3(kG / 32, kN / 64), blk, 0, stream>>>(
          th, grT + (size_t)h * kG * kG, nullptr, nullptr, inter, inter, nullptr,
          nullptr, kN, kG, kG);
  }
  k_final_add<<<cdiv(NG, 256), blk, 0, stream>>>(out_fin, out_sm, inter, (int)NG);
}
